// Adj_layer_34299608826046
// MI455X (gfx1250) — compile-verified, matrix operand order checked by reading, not run on hardware
//
#include <hip/hip_runtime.h>
#include <cfloat>
#include <cmath>

// ---------------------------------------------------------------- constants
#define BB    8
#define VV    201
#define DD    256
#define HH    128
#define KTOP  100
#define NPIX  (BB*VV*VV)       // 323208
#define NPAD  323328           // ceil(NPIX/128)*128
#define SLOPE 0.01f
#define EPSBN 1e-5f

typedef __attribute__((ext_vector_type(16))) _Float16 v16h;
typedef __attribute__((ext_vector_type(8)))  _Float16 v8h;
typedef __attribute__((ext_vector_type(8)))  float    v8f;

union F16x16 { v16h v; v8h h[2]; _Float16 e[16]; };

// ---------------------------------------------------------------- init / convert
__global__ void init_kernel(float* stats, float* sc0, float* sh0) {
    int i = blockIdx.x * blockDim.x + threadIdx.x;
    if (i < 2048) stats[i] = 0.f;          // 4 layers x (sum[256]+sumsq[256])
    if (i < 256)  { sc0[i] = 1.f; sh0[i] = 0.f; }
}

__global__ void f32_to_f16(const float* __restrict__ in, _Float16* __restrict__ out, int n) {
    int i = blockIdx.x * blockDim.x + threadIdx.x;
    if (i < n) out[i] = (_Float16)in[i];
}

// ---------------------------------------------------------------- |xi - xj|
__global__ void absdiff_kernel(const float* __restrict__ x, _Float16* __restrict__ out) {
    int n = blockIdx.x;                     // pixel id
    int c = threadIdx.x;                    // channel (256)
    int b   = n / (VV * VV);
    int rem = n - b * (VV * VV);
    int v   = rem / VV;
    int w   = rem - v * VV;
    float d = fabsf(x[((size_t)b * VV + v) * DD + c] - x[((size_t)b * VV + w) * DD + c]);
    out[(size_t)n * DD + c] = (_Float16)d;
}

// ---------------------------------------------------------------- fused GEMM + BN-stats
// in:  raw pre-activation of previous layer (f16) + per-channel (scale,shift)
//      -> leaky(scale*a+shift) applied while building A fragments
// out: raw pre-activation y = act @ W^T + b (f16), plus per-channel sum / sumsq
template<int CIN, int COUT>
__global__ __launch_bounds__(256)
void gemm_bn_stats(const _Float16* __restrict__ act,
                   const float*    __restrict__ inScale,
                   const float*    __restrict__ inShift,
                   const _Float16* __restrict__ Wh,      // [COUT][CIN] row-major f16
                   const float*    __restrict__ bias,    // [COUT]
                   _Float16*       __restrict__ out,     // [NPIX][COUT]
                   float*          __restrict__ gsum,
                   float*          __restrict__ gsumsq,
                   int Npix)
{
    constexpr int WAVES_N = COUT / 128;     // 2 (COUT=256) or 1 (COUT=128)
    constexpr int WAVES_M = 8 / WAVES_N;    // 4 or 8
    constexpr int KF      = CIN / 32;       // 8 or 4
    constexpr int NTILES  = 8;              // 8 x 16 = 128 cols per wave

    __shared__ float ssum[COUT];
    __shared__ float ssq [COUT];
    int tid = threadIdx.x;
    if (tid < COUT) { ssum[tid] = 0.f; ssq[tid] = 0.f; }
    __syncthreads();

    int wave = tid >> 5;
    int lane = tid & 31;
    int hh   = lane >> 4;     // lane half (K swizzle)
    int m    = lane & 15;

    int waveM   = wave / WAVES_N;
    int waveN   = wave % WAVES_N;
    int rowBase = (blockIdx.x * WAVES_M + waveM) * 16;
    int colBase = waveN * 128;

    // ---- A fragments for all K, activation applied on the fly --------------
    F16x16 afrag[KF];
    int  arow  = rowBase + m;
    bool rowOK = (arow < Npix);
    const _Float16* arowp = act + (size_t)arow * CIN;
    #pragma unroll
    for (int kk = 0; kk < KF; ++kk) {
        int c0 = kk * 32 + hh * 8;   // ISA 16-bit A layout: two contiguous 8-elt chunks
        int c1 = c0 + 16;
        if (rowOK) {
            v8h r0 = *reinterpret_cast<const v8h*>(arowp + c0);
            v8h r1 = *reinterpret_cast<const v8h*>(arowp + c1);
            #pragma unroll
            for (int e = 0; e < 8; ++e) {
                float a0 = (float)r0[e] * inScale[c0 + e] + inShift[c0 + e];
                float a1 = (float)r1[e] * inScale[c1 + e] + inShift[c1 + e];
                a0 = a0 >= 0.f ? a0 : SLOPE * a0;
                a1 = a1 >= 0.f ? a1 : SLOPE * a1;
                afrag[kk].e[e]     = (_Float16)a0;
                afrag[kk].e[e + 8] = (_Float16)a1;
            }
        } else {
            #pragma unroll
            for (int e = 0; e < 16; ++e) afrag[kk].e[e] = (_Float16)0.f;
        }
    }

    // ---- 8 N-tiles of 16 columns ------------------------------------------
    #pragma unroll
    for (int t = 0; t < NTILES; ++t) {
        int col = colBase + t * 16 + m;                 // B column == C column for this lane
        const _Float16* wrow = Wh + (size_t)col * CIN;  // W[col][*]  (B = W^T)
        v8f acc = {0.f,0.f,0.f,0.f,0.f,0.f,0.f,0.f};
        #pragma unroll
        for (int kk = 0; kk < KF; ++kk) {
            F16x16 bfr;
            int c0 = kk * 32 + hh * 8;
            bfr.h[0] = *reinterpret_cast<const v8h*>(wrow + c0);
            bfr.h[1] = *reinterpret_cast<const v8h*>(wrow + c0 + 16);
            acc = __builtin_amdgcn_wmma_f32_16x16x32_f16(
                      false, afrag[kk].v, false, bfr.v, (short)0, acc, false, false);
        }
        float bcol = bias[col];
        float s = 0.f, q = 0.f;
        #pragma unroll
        for (int r = 0; r < 8; ++r) {                   // C layout: row = r + hh*8
            int row = rowBase + r + hh * 8;
            if (row < Npix) {
                float y = acc[r] + bcol;
                out[(size_t)row * COUT + col] = (_Float16)y;
                s += y; q += y * y;
            }
        }
        atomicAdd(&ssum[col], s);
        atomicAdd(&ssq [col], q);
    }
    __syncthreads();
    if (tid < COUT) {
        atomicAdd(&gsum  [tid], ssum[tid]);
        atomicAdd(&gsumsq[tid], ssq [tid]);
    }
}

// ---------------------------------------------------------------- fold BN stats
__global__ void finalize_stats(const float* __restrict__ gsum, const float* __restrict__ gsumsq,
                               const float* __restrict__ g, const float* __restrict__ be,
                               float* __restrict__ outScale, float* __restrict__ outShift,
                               int C, float invN) {
    int c = threadIdx.x;
    if (c < C) {
        float mean = gsum[c] * invN;
        float var  = gsumsq[c] * invN - mean * mean;   // population var (ddof=0)
        float sc   = g[c] * rsqrtf(var + EPSBN);
        outScale[c] = sc;
        outShift[c] = be[c] - mean * sc;
    }
}

// ---------------------------------------------------------------- head: leaky(BN(y3)) . w4 + b4
__global__ void logits_kernel(const _Float16* __restrict__ y3,
                              const float* __restrict__ sc, const float* __restrict__ sh,
                              const float* __restrict__ w4, const float* __restrict__ b4,
                              float* __restrict__ logit, int Npix) {
    int n = blockIdx.x * blockDim.x + threadIdx.x;
    if (n >= Npix) return;
    const _Float16* p = y3 + (size_t)n * HH;
    float acc = b4[0];
    #pragma unroll 4
    for (int c = 0; c < HH; ++c) {
        float a = (float)p[c] * sc[c] + sh[c];
        a = a >= 0.f ? a : SLOPE * a;
        acc += a * w4[c];
    }
    logit[n] = acc;
}

// ---------------------------------------------------------------- softmax + top-K mask
__global__ __launch_bounds__(256)
void softmax_topk(const float* __restrict__ logits, float* __restrict__ out) {
    int row = blockIdx.x;                    // 0 .. B*V-1
    int tid = threadIdx.x;
    __shared__ float red[256];
    __shared__ float p  [256];

    float v = (tid < VV) ? logits[(size_t)row * VV + tid] : -FLT_MAX;
    red[tid] = v; __syncthreads();
    for (int s = 128; s > 0; s >>= 1) { if (tid < s) red[tid] = fmaxf(red[tid], red[tid + s]); __syncthreads(); }
    float mx = red[0]; __syncthreads();

    float e = (tid < VV) ? expf(v - mx) : 0.f;
    red[tid] = e; __syncthreads();
    for (int s = 128; s > 0; s >>= 1) { if (tid < s) red[tid] += red[tid + s]; __syncthreads(); }
    float inv = 1.f / red[0]; __syncthreads();

    float prob = e * inv;
    p[tid] = prob; __syncthreads();

    if (tid < VV) {
        int rank = 0;
        for (int i = 0; i < VV; ++i) {
            float pi = p[i];
            rank += (pi > prob) || (pi == prob && i < tid);   // top_k tie order: lower index first
        }
        out[(size_t)row * VV + tid] = (rank < KTOP) ? prob : 0.f;
    }
}

// ---------------------------------------------------------------- launch
extern "C" void kernel_launch(void* const* d_in, const int* in_sizes, int n_in,
                              void* d_out, int out_size, void* d_ws, size_t ws_size,
                              hipStream_t stream) {
    const float* x   = (const float*)d_in[0];
    const float* w0  = (const float*)d_in[1];
    const float* b0  = (const float*)d_in[2];
    const float* g0  = (const float*)d_in[3];
    const float* be0 = (const float*)d_in[4];
    const float* w1  = (const float*)d_in[5];
    const float* b1  = (const float*)d_in[6];
    const float* g1  = (const float*)d_in[7];
    const float* be1 = (const float*)d_in[8];
    const float* w2  = (const float*)d_in[9];
    const float* b2  = (const float*)d_in[10];
    const float* g2  = (const float*)d_in[11];
    const float* be2 = (const float*)d_in[12];
    const float* w3  = (const float*)d_in[13];
    const float* b3  = (const float*)d_in[14];
    const float* g3  = (const float*)d_in[15];
    const float* be3 = (const float*)d_in[16];
    const float* w4  = (const float*)d_in[17];
    const float* b4  = (const float*)d_in[18];

    char* ws = (char*)d_ws;
    size_t off = 0;
    auto take = [&](size_t bytes) { size_t o = off; off += (bytes + 255) & ~(size_t)255; return o; };

    _Float16* actA  = (_Float16*)(ws + take((size_t)NPAD * 256 * 2));
    _Float16* actB  = (_Float16*)(ws + take((size_t)NPAD * 256 * 2));
    _Float16* wh0   = (_Float16*)(ws + take(256 * 256 * 2));
    _Float16* wh1   = (_Float16*)(ws + take(256 * 256 * 2));
    _Float16* wh2   = (_Float16*)(ws + take(128 * 256 * 2));
    _Float16* wh3   = (_Float16*)(ws + take(128 * 128 * 2));
    float*    scbuf = (float*)   (ws + take(5 * 512 * 4));   // 5 stages x (scale[256],shift[256])
    float*    stats = (float*)   (ws + take(4 * 512 * 4));   // 4 layers x (sum[256],sumsq[256])
    float*    logit = (float*)   (ws + take((size_t)NPIX * 4));

    auto SC = [&](int s) { return scbuf + s * 512; };
    auto SH = [&](int s) { return scbuf + s * 512 + 256; };
    auto SUM = [&](int l) { return stats + l * 512; };
    auto SSQ = [&](int l) { return stats + l * 512 + 256; };

    const float invN = 1.f / (float)NPIX;

    init_kernel<<<8, 256, 0, stream>>>(stats, SC(0), SH(0));
    f32_to_f16<<<(65536 + 255) / 256, 256, 0, stream>>>(w0, wh0, 65536);
    f32_to_f16<<<(65536 + 255) / 256, 256, 0, stream>>>(w1, wh1, 65536);
    f32_to_f16<<<(32768 + 255) / 256, 256, 0, stream>>>(w2, wh2, 32768);
    f32_to_f16<<<(16384 + 255) / 256, 256, 0, stream>>>(w3, wh3, 16384);

    absdiff_kernel<<<NPIX, 256, 0, stream>>>(x, actA);

    gemm_bn_stats<256, 256><<<(NPIX + 63) / 64, 256, 0, stream>>>(
        actA, SC(0), SH(0), wh0, b0, actB, SUM(0), SSQ(0), NPIX);
    finalize_stats<<<1, 256, 0, stream>>>(SUM(0), SSQ(0), g0, be0, SC(1), SH(1), 256, invN);

    gemm_bn_stats<256, 256><<<(NPIX + 63) / 64, 256, 0, stream>>>(
        actB, SC(1), SH(1), wh1, b1, actA, SUM(1), SSQ(1), NPIX);
    finalize_stats<<<1, 256, 0, stream>>>(SUM(1), SSQ(1), g1, be1, SC(2), SH(2), 256, invN);

    gemm_bn_stats<256, 128><<<(NPIX + 127) / 128, 256, 0, stream>>>(
        actA, SC(2), SH(2), wh2, b2, actB, SUM(2), SSQ(2), NPIX);
    finalize_stats<<<1, 256, 0, stream>>>(SUM(2), SSQ(2), g2, be2, SC(3), SH(3), 128, invN);

    gemm_bn_stats<128, 128><<<(NPIX + 127) / 128, 256, 0, stream>>>(
        actB, SC(3), SH(3), wh3, b3, actA, SUM(3), SSQ(3), NPIX);
    finalize_stats<<<1, 256, 0, stream>>>(SUM(3), SSQ(3), g3, be3, SC(4), SH(4), 128, invN);

    logits_kernel<<<(NPIX + 255) / 256, 256, 0, stream>>>(actA, SC(4), SH(4), w4, b4, logit, NPIX);

    softmax_topk<<<BB * VV, 256, 0, stream>>>(logit, (float*)d_out);
}
